// LowRankKernel_86766929314310
// MI455X (gfx1250) — matrix operand order repaired and checked by reading, not compiled, not run on hardware
//
#include <hip/hip_runtime.h>
#include <hip/hip_fp16.h>

#define P_TOT   16384
#define HIDN    256
#define BATCH   64
#define CIN     64
#define COUT    64
#define RANKN   64
#define KSPLIT  8
#define KCH     (P_TOT / KSPLIT)   // 2048

typedef __attribute__((ext_vector_type(16))) _Float16 v16h;
typedef __attribute__((ext_vector_type(8)))  _Float16 v8h;
typedef __attribute__((ext_vector_type(8)))  float    v8f;
typedef __attribute__((ext_vector_type(4)))  float    v4f;
typedef __attribute__((ext_vector_type(4)))  unsigned int u32x4;
typedef __attribute__((ext_vector_type(8)))  int      i32x8;
typedef __attribute__((ext_vector_type(4)))  int      i32x4;

__device__ __forceinline__ v16h mk16(v8h lo, v8h hi) {
    v16h r;
#pragma unroll
    for (int e = 0; e < 8; ++e) { r[e] = lo[e]; r[8 + e] = hi[e]; }
    return r;
}

__device__ __forceinline__ v8f wmma_f16(v16h a, v16h b, v8f c) {
    // D = A(16x32 f16) * B(32x16 f16) + C(16x16 f32)
    return __builtin_amdgcn_wmma_f32_16x16x32_f16(
        false, a, false, b, (short)0, c, false, false);
}

__device__ __forceinline__ float gelu_exact(float x) {
    return 0.5f * x * (1.0f + erff(x * 0.70710678118654752f));
}

// ---------------------------------------------------------------- zero scratch
__global__ void k_zero(float* __restrict__ p, int n) {
    int i = blockIdx.x * 256 + threadIdx.x;
    if (i < n) p[i] = 0.0f;
}

// ------------------------------------------------- hidden activations (f16)
// hidp_t: [HIDN][P_TOT] (transposed, B-operand of T-GEMM)
// hidf  : [P_TOT][HIDN] (row-major, A-operand of out-GEMM)
__global__ __launch_bounds__(256) void k_hidden(
    const float* __restrict__ coords,
    const float* __restrict__ w1p, const float* __restrict__ b1p,
    const float* __restrict__ w1f, const float* __restrict__ b1f,
    _Float16* __restrict__ hidp_t, _Float16* __restrict__ hidf) {
    const int p = blockIdx.x;
    const int h = threadIdx.x;            // 0..255
    const float c0 = coords[2 * p];
    const float c1 = coords[2 * p + 1];
    float xp = fmaf(c0, w1p[h], fmaf(c1, w1p[HIDN + h], b1p[h]));
    float xf = fmaf(c0, w1f[h], fmaf(c1, w1f[HIDN + h], b1f[h]));
    hidp_t[(size_t)h * P_TOT + p] = (_Float16)gelu_exact(xp);
    hidf[(size_t)p * HIDN + h]    = (_Float16)gelu_exact(xf);
}

// -------------------------------------------- T[b,h,i] = sum_p v[b,i,p]*hid[p,h]
__global__ __launch_bounds__(256) void k_tmat(
    const float* __restrict__ v,          // [B][CIN][P]
    const _Float16* __restrict__ hidp_t,  // [HIDN][P]
    float* __restrict__ T,                // [B][HIDN][CIN]
    float* __restrict__ vsum) {           // [B][CIN]
    __shared__ __align__(16) _Float16 ldsA[BATCH * 32];   // [b][k]
    __shared__ __align__(16) _Float16 ldsB[HIDN * 32];    // [h][k]
    __shared__ float ldsR[BATCH * 4];

    const int i    = blockIdx.x;
    const int s    = blockIdx.y;
    const int t    = threadIdx.x;
    const int wave = t >> 5;
    const int lane = t & 31;
    const int m0   = (wave & 3) * 16;     // batch block
    const int n0   = (wave >> 2) * 128;   // hidden block

    v8f acc[8];
#pragma unroll
    for (int j = 0; j < 8; ++j)
#pragma unroll
        for (int e = 0; e < 8; ++e) acc[j][e] = 0.0f;

    const int bA = t >> 2;                // 0..63
    const int pA = (t & 3) * 8;           // 0,8,16,24
    const float*    vrow = v + ((size_t)bA * CIN + i) * P_TOT + (size_t)s * KCH;
    const _Float16* hrow = hidp_t + (size_t)t * P_TOT + (size_t)s * KCH;

    const int arow = m0 + (lane & 15);
    const int kbA  = (lane >> 4) * 8;
    const int kbB  = (lane >> 4) * 8;

    float vacc = 0.0f;

    for (int kt = 0; kt < KCH / 32; ++kt) {
        const int p0 = kt * 32;
        // prefetch the streaming operands ahead (v 8, hidden 32 iterations)
        __builtin_prefetch(vrow + p0 + pA + 256, 0, 1);
        __builtin_prefetch(hrow + p0 + 1024, 0, 1);
        // stage A: v (f32 -> f16)
        v4f a0 = *(const v4f*)(vrow + p0 + pA);
        v4f a1 = *(const v4f*)(vrow + p0 + pA + 4);
        vacc += a0[0] + a0[1] + a0[2] + a0[3] + a1[0] + a1[1] + a1[2] + a1[3];
        v8h ah;
#pragma unroll
        for (int e = 0; e < 4; ++e) { ah[e] = (_Float16)a0[e]; ah[4 + e] = (_Float16)a1[e]; }
        __syncthreads();
        *(v8h*)(ldsA + bA * 32 + pA) = ah;
        // stage B: hidden, already [h][k]-contiguous per row
        *(v8h*)(ldsB + t * 32 + 0)  = *(const v8h*)(hrow + p0 + 0);
        *(v8h*)(ldsB + t * 32 + 8)  = *(const v8h*)(hrow + p0 + 8);
        *(v8h*)(ldsB + t * 32 + 16) = *(const v8h*)(hrow + p0 + 16);
        *(v8h*)(ldsB + t * 32 + 24) = *(const v8h*)(hrow + p0 + 24);
        __syncthreads();

        v16h afrag = mk16(*(const v8h*)(ldsA + arow * 32 + kbA),
                          *(const v8h*)(ldsA + arow * 32 + 16 + kbA));
#pragma unroll
        for (int j = 0; j < 8; ++j) {
            const int h = n0 + j * 16 + (lane & 15);
            v16h bfrag = mk16(*(const v8h*)(ldsB + h * 32 + kbB),
                              *(const v8h*)(ldsB + h * 32 + kbB + 16));
            acc[j] = wmma_f16(afrag, bfrag, acc[j]);
        }
    }

    // reduce sum_p v[b,i,p] (4 partials per b)
    __syncthreads();
    ldsR[bA * 4 + (t & 3)] = vacc;
    __syncthreads();
    if ((t & 3) == 0) {
        float s4 = ldsR[bA * 4] + ldsR[bA * 4 + 1] + ldsR[bA * 4 + 2] + ldsR[bA * 4 + 3];
        atomicAdd(&vsum[bA * CIN + i], s4);
    }

    // accumulate tile into T
#pragma unroll
    for (int j = 0; j < 8; ++j) {
        const int h    = n0 + j * 16 + (lane & 15);
        const int brow = m0 + 8 * (lane >> 4);
#pragma unroll
        for (int e = 0; e < 8; ++e)
            atomicAdd(&T[((size_t)(brow + e) * HIDN + h) * CIN + i], acc[j][e]);
    }
}

// ---------------- l2[b,r] = dx^2 * ( sum_{h,i} T[b,h,i] w2p[h,r*64+i] + sum_i vsum[b,i] b2p[r*64+i] )
__global__ __launch_bounds__(256) void k_l2(
    const float* __restrict__ T,       // [B][HIDN*CIN]
    const float* __restrict__ w2p,     // [HIDN][RANKN*CIN]
    const float* __restrict__ b2p,     // [RANKN*CIN]
    const float* __restrict__ vsum,    // [B][CIN]
    const float* __restrict__ coords,
    float* __restrict__ l2) {          // [B][RANKN]
    __shared__ float red[256 * 16];
    const int b  = blockIdx.x;
    const int rg = blockIdx.y;         // r-group of 16
    const int t  = threadIdx.x;
    float acc[16];
#pragma unroll
    for (int q = 0; q < 16; ++q) acc[q] = 0.0f;

    const float* Tb = T + (size_t)b * (HIDN * CIN);
    for (int k = t; k < HIDN * CIN; k += 256) {
        const float tv = Tb[k];
        const int h = k >> 6, i = k & 63;
        const float* wrow = w2p + (size_t)h * (RANKN * CIN) + i;
#pragma unroll
        for (int q = 0; q < 16; ++q)
            acc[q] = fmaf(tv, wrow[(rg * 16 + q) * 64], acc[q]);
    }
    if (t < 64) {  // layer-2 bias term through the integral
        const float vs = vsum[b * 64 + t];
#pragma unroll
        for (int q = 0; q < 16; ++q)
            acc[q] = fmaf(vs, b2p[(rg * 16 + q) * 64 + t], acc[q]);
    }
#pragma unroll
    for (int q = 0; q < 16; ++q) red[t * 16 + q] = acc[q];
    __syncthreads();
    for (int stride = 128; stride >= 1; stride >>= 1) {
        if (t < stride)
#pragma unroll
            for (int q = 0; q < 16; ++q) red[t * 16 + q] += red[(t + stride) * 16 + q];
        __syncthreads();
    }
    if (t < 16) {
        const float dx = coords[2] - coords[0];
        l2[b * RANKN + rg * 16 + t] = red[t] * dx * dx;
    }
}

// ------- S_t[b,o,h] = sum_r l2[b,r] w2f[h,r*64+o]; bias2[b,o] = sum_r l2[b,r] b2f[r*64+o]
__global__ __launch_bounds__(256) void k_smat(
    const float* __restrict__ l2,
    const float* __restrict__ w2f,     // [HIDN][RANKN*COUT]
    const float* __restrict__ b2f,     // [RANKN*COUT]
    _Float16* __restrict__ S,          // [B][COUT][HIDN]  (pre-transposed!)
    float* __restrict__ bias2) {       // [B][COUT]
    __shared__ float l2s[64];
    const int b  = blockIdx.x;
    const int hb = blockIdx.y;         // 8 blocks of 32 h-rows
    const int t  = threadIdx.x;
    if (t < 64) l2s[t] = l2[b * 64 + t];
    __syncthreads();
    const int h  = hb * 32 + (t >> 3);
    const int o0 = (t & 7) * 8;
    float acc[8];
#pragma unroll
    for (int j = 0; j < 8; ++j) acc[j] = 0.0f;
    const float* w = w2f + (size_t)h * 4096 + o0;
    for (int r = 0; r < 64; ++r) {
        const float lv = l2s[r];
#pragma unroll
        for (int j = 0; j < 8; ++j) acc[j] = fmaf(lv, w[r * 64 + j], acc[j]);
    }
#pragma unroll
    for (int j = 0; j < 8; ++j)
        S[((size_t)b * COUT + o0 + j) * HIDN + h] = (_Float16)acc[j];
    if (hb == 0 && t < 64) {
        float sb = 0.0f;
        for (int r = 0; r < 64; ++r) sb = fmaf(l2s[r], b2f[r * 64 + t], sb);
        bias2[b * 64 + t] = sb;
    }
}

// ------------- out[b,o,p] = sum_h hidf[p,h] * S_t[b,o,h] + bias2[b,o]
__global__ __launch_bounds__(256) void k_out(
    const _Float16* __restrict__ hidf,   // [P][HIDN]
    const _Float16* __restrict__ S,      // [B][COUT][HIDN]
    const float* __restrict__ bias2,     // [B][COUT]
    float* __restrict__ out) {           // [B][COUT][P]
    __shared__ __align__(16) _Float16 ldsS[COUT * HIDN];   // [o][h], 32 KB
    const int pt   = blockIdx.x;          // 128 tiles of 128 p
    const int b    = blockIdx.y;
    const int t    = threadIdx.x;
    const int wave = t >> 5;
    const int lane = t & 31;

    const _Float16* sbase = S + (size_t)b * (COUT * HIDN);  // 32 KB, contiguous

#if __has_builtin(__builtin_amdgcn_tensor_load_to_lds) && __has_builtin(__builtin_amdgcn_s_wait_tensorcnt)
    // Tensor Data Mover: 1-D DMA of the 32 KB S_b tile, global -> LDS.
    if (t < 32) {                         // wave 0 issues the TDM descriptor
        const unsigned long long ga = (unsigned long long)(uintptr_t)sbase;
        const unsigned int ldsoff = (unsigned int)(uintptr_t)(&ldsS[0]);
        u32x4 g0;
        g0[0] = 1u;                        // count = 1 valid descriptor
        g0[1] = ldsoff;                    // lds_addr
        g0[2] = (unsigned int)ga;          // global_addr[31:0]
        g0[3] = ((unsigned int)(ga >> 32) & 0x01FFFFFFu) | 0x80000000u; // addr[56:32] | type=2
        i32x8 g1;
        g1[0] = 0x00030000;                // data_size = 3 (8B units)
        g1[1] = 0x10000000;                // tensor_dim0 = 4096 (lo16 in [31:16])
        g1[2] = 0x00010000;                // tensor_dim1 = 1
        g1[3] = 0x10000000;                // tile_dim0 = 4096
        g1[4] = 0x00000001;                // tile_dim1 = 1, tile_dim2 = 0
        g1[5] = 4096;                      // tensor_dim0_stride = 4096
        g1[6] = 0;
        g1[7] = 0;
        i32x4 gz4 = {0, 0, 0, 0};
        i32x8 gz8 = {0, 0, 0, 0, 0, 0, 0, 0};
        __builtin_amdgcn_tensor_load_to_lds(g0, g1, gz4, gz4, gz8, 0);
        __builtin_amdgcn_s_wait_tensorcnt(0);
    }
#else
    // fallback: coalesced 32 KB copy, 16 B per thread per step
    for (int idx = t * 8; idx < COUT * HIDN; idx += 256 * 8)
        *(v8h*)(ldsS + idx) = *(const v8h*)(sbase + idx);
#endif
    __syncthreads();

    const int p0 = pt * 128 + wave * 16;
    v8f acc[4];
#pragma unroll
    for (int j = 0; j < 4; ++j)
#pragma unroll
        for (int e = 0; e < 8; ++e) acc[j][e] = 0.0f;

    const _Float16* arow = hidf + (size_t)(p0 + (lane & 15)) * HIDN;
    const int kbA = (lane >> 4) * 8;
    const int kbB = (lane >> 4) * 8;

#pragma unroll
    for (int kt = 0; kt < HIDN / 32; ++kt) {
        const int k0 = kt * 32;
        v16h afrag = mk16(*(const v8h*)(arow + k0 + kbA),
                          *(const v8h*)(arow + k0 + 16 + kbA));
#pragma unroll
        for (int j = 0; j < 4; ++j) {
            const int o = j * 16 + (lane & 15);
            v16h bfrag = mk16(*(const v8h*)(ldsS + o * HIDN + k0 + kbB),
                              *(const v8h*)(ldsS + o * HIDN + k0 + kbB + 16));
            acc[j] = wmma_f16(afrag, bfrag, acc[j]);
        }
    }

#pragma unroll
    for (int j = 0; j < 4; ++j) {
        const int o = j * 16 + (lane & 15);
        const float bb = bias2[b * 64 + o];
        float* orow = out + ((size_t)b * COUT + o) * P_TOT + p0 + 8 * (lane >> 4);
        v4f s0, s1;
#pragma unroll
        for (int e = 0; e < 4; ++e) { s0[e] = acc[j][e] + bb; s1[e] = acc[j][4 + e] + bb; }
        *(v4f*)(orow)     = s0;
        *(v4f*)(orow + 4) = s1;
    }
}

extern "C" void kernel_launch(void* const* d_in, const int* in_sizes, int n_in,
                              void* d_out, int out_size, void* d_ws, size_t ws_size,
                              hipStream_t stream) {
    const float* v      = (const float*)d_in[0];
    const float* coords = (const float*)d_in[1];
    const float* w1p    = (const float*)d_in[2];
    const float* b1p    = (const float*)d_in[3];
    const float* w2p    = (const float*)d_in[4];
    const float* b2p    = (const float*)d_in[5];
    const float* w1f    = (const float*)d_in[6];
    const float* b1f    = (const float*)d_in[7];
    const float* w2f    = (const float*)d_in[8];
    const float* b2f    = (const float*)d_in[9];
    float* out = (float*)d_out;

    char* ws = (char*)d_ws;
    const size_t MB = 1024ull * 1024ull;
    _Float16* hidp_t = (_Float16*)(ws);                    // 8 MB  [HIDN][P]
    _Float16* hidf   = (_Float16*)(ws + 8 * MB);           // 8 MB  [P][HIDN]
    float*    T      = (float*)(ws + 16 * MB);             // 4 MB  [B][HIDN][CIN]
    float*    vsum   = (float*)(ws + 20 * MB);             // 16 KB [B][CIN]
    float*    l2     = (float*)(ws + 20 * MB + 16 * 1024); // 16 KB [B][RANKN]
    float*    bias2  = (float*)(ws + 20 * MB + 32 * 1024); // 16 KB [B][COUT]
    _Float16* S      = (_Float16*)(ws + 20 * MB + 64 * 1024); // 2 MB [B][COUT][HIDN]

    // zero T/vsum/l2/bias2 (contiguous region)
    const int nzero = (4 * MB + 48 * 1024) / 4;
    k_zero<<<(nzero + 255) / 256, 256, 0, stream>>>(T, nzero);

    k_hidden<<<P_TOT, 256, 0, stream>>>(coords, w1p, b1p, w1f, b1f, hidp_t, hidf);
    k_tmat<<<dim3(CIN, KSPLIT), 256, 0, stream>>>(v, hidp_t, T, vsum);
    k_l2<<<dim3(BATCH, 4), 256, 0, stream>>>(T, w2p, b2p, vsum, coords, l2);
    k_smat<<<dim3(BATCH, 8), 256, 0, stream>>>(l2, w2f, b2f, S, bias2);
    k_out<<<dim3(P_TOT / 128, BATCH), 256, 0, stream>>>(hidf, S, bias2, out);
}
